// Attention_71975061946902
// MI455X (gfx1250) — matrix operand order repaired and checked by reading, not compiled, not run on hardware
//
#include <hip/hip_runtime.h>
#include <hip/hip_bf16.h>
#include <stdint.h>

// Problem constants (fixed by the reference)
#define BATCH  2
#define NB     1024
#define EB     2048
#define LTOT   3072          // N + E
#define HEADS  8
#define DHEAD  64
#define INNER  512           // HEADS * DHEAD
#define SCALE  0.125f        // DHEAD^-0.5

typedef __attribute__((ext_vector_type(16))) __bf16 bf16x16;
typedef __attribute__((ext_vector_type(2)))  __bf16 bf16x2;
typedef __attribute__((ext_vector_type(8)))  float  floatx8;

union FragBF { bf16x16 v; unsigned int d[8]; unsigned short u[16]; };
union FragF  { floatx8 v; float f[8]; };

// hardware float->bf16 convert (let the backend pick v_cvt_*bf16*)
__device__ __forceinline__ unsigned short f2bf(float x) {
  union { __bf16 h; unsigned short u; } c; c.h = (__bf16)x;
  return c.u;
}
__device__ __forceinline__ unsigned pack2bf(float a, float b) {
  union { bf16x2 h; unsigned u; } c;
  c.h.x = (__bf16)a; c.h.y = (__bf16)b;
  return c.u;
}
// Two 16-byte loads -> one 16x32 (A) or 32x16 (B) bf16 fragment
__device__ __forceinline__ void load_frag_b128(FragBF& fr,
                                               const unsigned short* lo_p,
                                               const unsigned short* hi_p) {
  uint4 a = *(const uint4*)lo_p;
  uint4 b = *(const uint4*)hi_p;
  fr.d[0] = a.x; fr.d[1] = a.y; fr.d[2] = a.z; fr.d[3] = a.w;
  fr.d[4] = b.x; fr.d[5] = b.y; fr.d[6] = b.z; fr.d[7] = b.w;
}

// ---------------------------------------------------------------------------
// One-shot preparation kernels (outputs are L2-resident and reused many times)
// ---------------------------------------------------------------------------
__global__ void maskbias_kernel(const unsigned char* __restrict__ mask,
                                float* __restrict__ mb, int n) {
  int i = blockIdx.x * blockDim.x + threadIdx.x;
  if (i < n) mb[i] = mask[i] ? 0.0f : -1.0e30f;
}

// fp32 -> bf16, 4 elements / thread
__global__ void cvt_bf16_kernel(const float* __restrict__ in,
                                unsigned short* __restrict__ out, int n) {
  int i = (blockIdx.x * blockDim.x + threadIdx.x) * 4;
  if (i < n) {
    float4 v = *(const float4*)(in + i);
    uint2 p; p.x = pack2bf(v.x, v.y); p.y = pack2bf(v.z, v.w);
    *(uint2*)(out + i) = p;
  }
}

// fp32 W[din][dout] -> bf16 Wt[dout][din]
__global__ void cvt_transpose_kernel(const float* __restrict__ W,
                                     unsigned short* __restrict__ Wt,
                                     int din, int dout) {
  int i = blockIdx.x * blockDim.x + threadIdx.x;
  if (i < din * dout) {
    int r = i / dout, c = i - r * dout;
    Wt[c * din + r] = f2bf(W[i]);
  }
}

// ---------------------------------------------------------------------------
// 64x64 tiled WMMA GEMM:  Out = A(rows x din) * Wt^T + bias
// A: bf16 row-major; Wt: bf16 transposed [dout][din].
// Both tiles staged with async global->LDS b128 copies; fragments are b128.
//   out_mode 0: bf16 QKV layout  [((b*H+h)*LTOT + out_lbase + r)*64 + d]
//   out_mode 1: fp32 row-major   [b*rows*dout + r*dout + c]
//   out_mode 2: bf16 transposed  [((b*H+h)*DHEAD + d)*LTOT + out_lbase + r]
// ---------------------------------------------------------------------------
#define A_STRIDE 40   // 80 B rows: 16B-aligned, bank-conflict-free
#define BT_STRIDE 40

__global__ void __launch_bounds__(256) gemm_wmma_kernel(
    const unsigned short* __restrict__ A, int a_lbase, int a_ltot,
    const unsigned short* __restrict__ Wt, const float* __restrict__ bias,
    void* __restrict__ Out, int out_mode, int out_lbase,
    int rows, int din, int dout)
{
  __shared__ __align__(16) unsigned short Abuf[64 * A_STRIDE];
  __shared__ __align__(16) unsigned short Bt[64 * BT_STRIDE];   // [col][k]

  const int t    = threadIdx.x;
  const int lane = t & 31, wid = t >> 5;
  const int wm   = wid & 3, wn = wid >> 2;
  const int rowtile = blockIdx.x, coltile = blockIdx.y, b = blockIdx.z;
  const int m16 = lane & 15, khalf = lane >> 4;

  FragF acc[2];
#pragma unroll
  for (int c2 = 0; c2 < 2; ++c2)
#pragma unroll
    for (int i = 0; i < 8; ++i) acc[c2].f[i] = 0.f;

  const int srow = t >> 2, scb = (t & 3) << 3;       // staging map: 8 elems/thread
  const unsigned short* Abase =
      A + ((size_t)b * a_ltot + a_lbase + rowtile * 64 + srow) * din + scb;
  const unsigned short* Wbase = Wt + ((size_t)coltile * 64 + srow) * din + scb;
  const unsigned adst = (unsigned)(uintptr_t)&Abuf[srow * A_STRIDE + scb];
  const unsigned bdst = (unsigned)(uintptr_t)&Bt[srow * BT_STRIDE + scb];

  const int ksteps = din >> 5;
  for (int ks = 0; ks < ksteps; ++ks) {
    const int k0 = ks << 5;
    asm volatile("global_load_async_to_lds_b128 %0, %1, off"
                 :: "v"(adst), "v"((unsigned long long)(uintptr_t)(Abase + k0))
                 : "memory");
    asm volatile("global_load_async_to_lds_b128 %0, %1, off"
                 :: "v"(bdst), "v"((unsigned long long)(uintptr_t)(Wbase + k0))
                 : "memory");
    asm volatile("s_wait_asynccnt 0x0" ::: "memory");
    __syncthreads();

    FragBF af;   // A fragment: two b128 loads
    {
      const unsigned short* p0 = &Abuf[(wm * 16 + m16) * A_STRIDE + 8 * khalf];
      load_frag_b128(af, p0, p0 + 16);
    }
#pragma unroll
    for (int c2 = 0; c2 < 2; ++c2) {
      FragBF bf;   // B fragment: k = 8*khalf + 0..7, 16 + 8*khalf + 0..7 in Bt row
      const unsigned short* p = &Bt[(wn * 32 + c2 * 16 + m16) * BT_STRIDE + 8 * khalf];
      load_frag_b128(bf, p, p + 16);
      acc[c2].v = __builtin_amdgcn_wmma_f32_16x16x32_bf16(
          false, af.v, false, bf.v, (short)0, acc[c2].v, false, false);
    }
    __syncthreads();
  }

  // ---- epilogue ----
#pragma unroll
  for (int c2 = 0; c2 < 2; ++c2) {
    int c = coltile * 64 + wn * 32 + c2 * 16 + m16;
    float bv = bias[c];
#pragma unroll
    for (int v8 = 0; v8 < 8; ++v8) {
      int r = rowtile * 64 + wm * 16 + v8 + 8 * khalf;   // C-layout row
      float val = acc[c2].f[v8] + bv;
      if (out_mode == 0) {
        int h = c >> 6, d = c & 63;
        unsigned short* o = (unsigned short*)Out;
        o[(((size_t)b * HEADS + h) * LTOT + out_lbase + r) * DHEAD + d] = f2bf(val);
      } else if (out_mode == 1) {
        float* o = (float*)Out;
        o[(size_t)b * rows * dout + (size_t)r * dout + c] = val;
      } else {  // transposed bf16 (V): [B,H,D,L]
        int h = c >> 6, d = c & 63;
        unsigned short* o = (unsigned short*)Out;
        o[(((size_t)b * HEADS + h) * DHEAD + d) * LTOT + out_lbase + r] = f2bf(val);
      }
    }
  }
}

// ---------------------------------------------------------------------------
// Flash attention: grid (LTOT/64, H, B), 128 threads = 4 waves x 16 Q rows.
// K tile LDS: Kt[n][k]; V tile LDS: VtT[d][k] (global V is [B,H,D,L]).
// K/V double-buffered via async global->LDS copies: tile j+1 is in flight
// while tile j computes; s_wait_asynccnt 8 drains only the older batch.
// ---------------------------------------------------------------------------
#define KV_STRIDE 72   // 144 B rows: 16B-aligned, bank-conflict-free

__global__ void __launch_bounds__(128) flash_attn_kernel(
    const unsigned short* __restrict__ Q,
    const unsigned short* __restrict__ K,
    const unsigned short* __restrict__ Vg,   // transposed [B,H,D,L]
    const float* __restrict__ mbias,
    unsigned short* __restrict__ O)
{
  __shared__ __align__(16) unsigned short Kt[2][64 * KV_STRIDE];
  __shared__ __align__(16) unsigned short VtT[2][64 * KV_STRIDE];
  __shared__ __align__(16) unsigned short Pb[4][16 * KV_STRIDE];
  __shared__ __align__(16) float mball[LTOT];

  const int t = threadIdx.x;
  const int lane = t & 31, wave = t >> 5;
  const int qtile = blockIdx.x, h = blockIdx.y, b = blockIdx.z;
  const int rowbase = qtile * 64 + wave * 16;
  const size_t bh = ((size_t)b * HEADS + h) * LTOT;
  const int m16 = lane & 15, khalf = lane >> 4;
  const int ntiles = LTOT / 64;

  // staging map: each thread owns half of one 64-elem row
  const int sr = t >> 1, shalf = t & 1;
  const unsigned short* kbase = K + (bh + sr) * DHEAD + shalf * 32;
  const unsigned short* vbase =
      Vg + (((size_t)b * HEADS + h) * DHEAD + sr) * LTOT + shalf * 32;

  auto issue_tile = [&](int jj, int buf) {
    unsigned kdst = (unsigned)(uintptr_t)&Kt[buf][sr * KV_STRIDE + shalf * 32];
    unsigned vdst = (unsigned)(uintptr_t)&VtT[buf][sr * KV_STRIDE + shalf * 32];
    const unsigned short* ksrc = kbase + (size_t)jj * 64 * DHEAD;
    const unsigned short* vsrc = vbase + (size_t)jj * 64;
#pragma unroll
    for (int i = 0; i < 4; ++i) {
      asm volatile("global_load_async_to_lds_b128 %0, %1, off"
                   :: "v"(kdst + i * 16),
                      "v"((unsigned long long)(uintptr_t)(ksrc + i * 8))
                   : "memory");
      asm volatile("global_load_async_to_lds_b128 %0, %1, off"
                   :: "v"(vdst + i * 16),
                      "v"((unsigned long long)(uintptr_t)(vsrc + i * 8))
                   : "memory");
    }
  };

  // hoist the whole mask-bias row for this batch into LDS (once)
  for (int i = t; i < LTOT / 4; i += 128)
    *(float4*)&mball[i * 4] = *(const float4*)(mbias + (size_t)b * LTOT + i * 4);

  // Q fragments: 16x64 = two 16x32 A fragments, b128 global loads, in regs
  FragBF qf[2];
  {
    const unsigned short* qp = Q + (bh + rowbase + m16) * DHEAD;
#pragma unroll
    for (int kk = 0; kk < 2; ++kk) {
      const unsigned short* p = qp + kk * 32 + 8 * khalf;
      load_frag_b128(qf[kk], p, p + 16);
    }
  }

  float mrow[8], lrow[8];
  FragF oacc[4];
#pragma unroll
  for (int i = 0; i < 8; ++i) { mrow[i] = -3.0e38f; lrow[i] = 0.f; }
#pragma unroll
  for (int d = 0; d < 4; ++d)
#pragma unroll
    for (int i = 0; i < 8; ++i) oacc[d].f[i] = 0.f;

  issue_tile(0, 0);
  int cur = 0;
  for (int j = 0; j < ntiles; ++j) {
    __syncthreads();                       // prev compute done: next buf free
    int jn = j + 1 == ntiles ? 0 : j + 1;  // wrap issue keeps wait imm constant
    issue_tile(jn, cur ^ 1);
    asm volatile("s_wait_asynccnt 0x8" ::: "memory");   // batch j landed
    __syncthreads();

    // ---- S = Q * K^T (4 chunks of 16 cols) ----
    FragF s[4];
#pragma unroll
    for (int nc = 0; nc < 4; ++nc) {
      floatx8 a = {0.f, 0.f, 0.f, 0.f, 0.f, 0.f, 0.f, 0.f};
#pragma unroll
      for (int kk = 0; kk < 2; ++kk) {
        FragBF bf;   // B(k,n)=K[n][k]: two 8-elem runs in Kt row
        const unsigned short* p =
            &Kt[cur][(nc * 16 + m16) * KV_STRIDE + kk * 32 + 8 * khalf];
        load_frag_b128(bf, p, p + 16);
        a = __builtin_amdgcn_wmma_f32_16x16x32_bf16(
            false, qf[kk].v, false, bf.v, (short)0, a, false, false);
      }
      s[nc].v = a;
      float mv = mball[j * 64 + nc * 16 + m16];
#pragma unroll
      for (int v8 = 0; v8 < 8; ++v8) s[nc].f[v8] = s[nc].f[v8] * SCALE + mv;
    }

    // ---- online softmax (row = v8 + 8*khalf; reduce across 16 lanes) ----
    float newm[8], corr[8], rsum[8];
#pragma unroll
    for (int v8 = 0; v8 < 8; ++v8) {
      float x = fmaxf(fmaxf(s[0].f[v8], s[1].f[v8]), fmaxf(s[2].f[v8], s[3].f[v8]));
      x = fmaxf(x, __shfl_xor(x, 1, 32));
      x = fmaxf(x, __shfl_xor(x, 2, 32));
      x = fmaxf(x, __shfl_xor(x, 4, 32));
      x = fmaxf(x, __shfl_xor(x, 8, 32));
      float nm = fmaxf(mrow[v8], x);
      newm[v8] = nm;
      corr[v8] = __expf(mrow[v8] - nm);
      mrow[v8] = nm;
      rsum[v8] = 0.f;
    }
#pragma unroll
    for (int nc = 0; nc < 4; ++nc)
#pragma unroll
      for (int v8 = 0; v8 < 8; ++v8) {
        float p = __expf(s[nc].f[v8] - newm[v8]);
        rsum[v8] += p;
        Pb[wave][(v8 + 8 * khalf) * KV_STRIDE + nc * 16 + m16] = f2bf(p);
      }
#pragma unroll
    for (int v8 = 0; v8 < 8; ++v8) {
      float x = rsum[v8];
      x += __shfl_xor(x, 1, 32);
      x += __shfl_xor(x, 2, 32);
      x += __shfl_xor(x, 4, 32);
      x += __shfl_xor(x, 8, 32);
      lrow[v8] = lrow[v8] * corr[v8] + x;
    }
#pragma unroll
    for (int d = 0; d < 4; ++d)
#pragma unroll
      for (int v8 = 0; v8 < 8; ++v8) oacc[d].f[v8] *= corr[v8];

    // per-wave P stores must land before A-fragment reads
    asm volatile("s_wait_dscnt 0x0" ::: "memory");

    // ---- O += P * V ----
#pragma unroll
    for (int dch = 0; dch < 4; ++dch)
#pragma unroll
      for (int kk = 0; kk < 2; ++kk) {
        FragBF pf;   // A fragment from per-wave P tile
        const unsigned short* pp =
            &Pb[wave][m16 * KV_STRIDE + kk * 32 + 8 * khalf];
        load_frag_b128(pf, pp, pp + 16);
        FragBF vf;   // B(k,d)=V[d][k]: two 8-elem runs in VtT row
        const unsigned short* vp =
            &VtT[cur][(dch * 16 + m16) * KV_STRIDE + kk * 32 + 8 * khalf];
        load_frag_b128(vf, vp, vp + 16);
        oacc[dch].v = __builtin_amdgcn_wmma_f32_16x16x32_bf16(
            false, pf.v, false, vf.v, (short)0, oacc[dch].v, false, false);
      }
    cur ^= 1;
  }

  // ---- normalize + write O [B, LTOT, INNER] bf16 ----
#pragma unroll
  for (int v8 = 0; v8 < 8; ++v8) {
    float inv = lrow[v8] > 0.f ? 1.f / lrow[v8] : 0.f;
    int row = rowbase + v8 + 8 * khalf;
    size_t base = ((size_t)b * LTOT + row) * INNER + (size_t)h * DHEAD;
#pragma unroll
    for (int dch = 0; dch < 4; ++dch)
      O[base + dch * 16 + m16] = f2bf(oacc[dch].f[v8] * inv);
  }
}

// ---------------------------------------------------------------------------
extern "C" void kernel_launch(void* const* d_in, const int* in_sizes, int n_in,
                              void* d_out, int out_size, void* d_ws, size_t ws_size,
                              hipStream_t stream) {
  const float* nodes = (const float*)d_in[0];
  const float* edges = (const float*)d_in[1];
  const unsigned char* mask = (const unsigned char*)d_in[2];
  const float* Wq  = (const float*)d_in[3];  const float* bq  = (const float*)d_in[4];
  const float* Wk  = (const float*)d_in[5];  const float* bk  = (const float*)d_in[6];
  const float* Wv  = (const float*)d_in[7];  const float* bv  = (const float*)d_in[8];
  const float* Weq = (const float*)d_in[9];  const float* beq = (const float*)d_in[10];
  const float* Wek = (const float*)d_in[11]; const float* bek = (const float*)d_in[12];
  const float* Wev = (const float*)d_in[13]; const float* bev = (const float*)d_in[14];
  const float* Wo  = (const float*)d_in[15]; const float* bo  = (const float*)d_in[16];
  const float* Weo = (const float*)d_in[17]; const float* beo = (const float*)d_in[18];

  char* ws = (char*)d_ws;
  size_t off = 0;
  auto alloc = [&](size_t bytes) { char* p = ws + off; off += bytes; return p; };
  const size_t qkvBytes = (size_t)BATCH * HEADS * LTOT * DHEAD * 2;   // 6 MB each
  unsigned short* Qb = (unsigned short*)alloc(qkvBytes);
  unsigned short* Kb = (unsigned short*)alloc(qkvBytes);
  unsigned short* Vb = (unsigned short*)alloc(qkvBytes);   // [B,H,D,L]
  unsigned short* Ob = (unsigned short*)alloc(qkvBytes);
  float* mbias       = (float*)alloc((size_t)BATCH * LTOT * 4);
  unsigned short* nodesBf = (unsigned short*)alloc((size_t)BATCH * NB * 128 * 2);
  unsigned short* edgesBf = (unsigned short*)alloc((size_t)BATCH * EB * 256 * 2);
  unsigned short* WqT  = (unsigned short*)alloc((size_t)128 * INNER * 2);
  unsigned short* WkT  = (unsigned short*)alloc((size_t)128 * INNER * 2);
  unsigned short* WvT  = (unsigned short*)alloc((size_t)128 * INNER * 2);
  unsigned short* WeqT = (unsigned short*)alloc((size_t)256 * INNER * 2);
  unsigned short* WekT = (unsigned short*)alloc((size_t)256 * INNER * 2);
  unsigned short* WevT = (unsigned short*)alloc((size_t)256 * INNER * 2);
  unsigned short* WoT  = (unsigned short*)alloc((size_t)INNER * 128 * 2);
  unsigned short* WeoT = (unsigned short*)alloc((size_t)INNER * 256 * 2);

  // ---- one-shot prep ----
  maskbias_kernel<<<dim3((BATCH * LTOT + 255) / 256), 256, 0, stream>>>(
      mask, mbias, BATCH * LTOT);
  cvt_bf16_kernel<<<dim3(BATCH * NB * 128 / 1024), 256, 0, stream>>>(
      nodes, nodesBf, BATCH * NB * 128);
  cvt_bf16_kernel<<<dim3(BATCH * EB * 256 / 1024), 256, 0, stream>>>(
      edges, edgesBf, BATCH * EB * 256);
  cvt_transpose_kernel<<<dim3(128 * INNER / 256), 256, 0, stream>>>(Wq,  WqT,  128, INNER);
  cvt_transpose_kernel<<<dim3(128 * INNER / 256), 256, 0, stream>>>(Wk,  WkT,  128, INNER);
  cvt_transpose_kernel<<<dim3(128 * INNER / 256), 256, 0, stream>>>(Wv,  WvT,  128, INNER);
  cvt_transpose_kernel<<<dim3(256 * INNER / 256), 256, 0, stream>>>(Weq, WeqT, 256, INNER);
  cvt_transpose_kernel<<<dim3(256 * INNER / 256), 256, 0, stream>>>(Wek, WekT, 256, INNER);
  cvt_transpose_kernel<<<dim3(256 * INNER / 256), 256, 0, stream>>>(Wev, WevT, 256, INNER);
  cvt_transpose_kernel<<<dim3(INNER * 128 / 256), 256, 0, stream>>>(Wo,  WoT,  INNER, 128);
  cvt_transpose_kernel<<<dim3(INNER * 256 / 256), 256, 0, stream>>>(Weo, WeoT, INNER, 256);

  // ---- QKV projections ----
  dim3 gN(NB / 64, INNER / 64, BATCH);
  gemm_wmma_kernel<<<gN, 256, 0, stream>>>(nodesBf, 0, NB, WqT, bq, Qb, 0, 0,  NB, 128, INNER);
  gemm_wmma_kernel<<<gN, 256, 0, stream>>>(nodesBf, 0, NB, WkT, bk, Kb, 0, 0,  NB, 128, INNER);
  gemm_wmma_kernel<<<gN, 256, 0, stream>>>(nodesBf, 0, NB, WvT, bv, Vb, 2, 0,  NB, 128, INNER);
  dim3 gE(EB / 64, INNER / 64, BATCH);
  gemm_wmma_kernel<<<gE, 256, 0, stream>>>(edgesBf, 0, EB, WeqT, beq, Qb, 0, NB, EB, 256, INNER);
  gemm_wmma_kernel<<<gE, 256, 0, stream>>>(edgesBf, 0, EB, WekT, bek, Kb, 0, NB, EB, 256, INNER);
  gemm_wmma_kernel<<<gE, 256, 0, stream>>>(edgesBf, 0, EB, WevT, bev, Vb, 2, NB, EB, 256, INNER);

  // ---- attention ----
  flash_attn_kernel<<<dim3(LTOT / 64, HEADS, BATCH), 128, 0, stream>>>(
      Qb, Kb, Vb, mbias, Ob);

  // ---- output projections (fp32 into d_out: node_out then edge_out) ----
  float* nodeOut = (float*)d_out;
  float* edgeOut = nodeOut + (size_t)BATCH * NB * 128;
  gemm_wmma_kernel<<<dim3(NB / 64, 128 / 64, BATCH), 256, 0, stream>>>(
      Ob, 0, LTOT, WoT, bo, nodeOut, 1, 0, NB, INNER, 128);
  gemm_wmma_kernel<<<dim3(EB / 64, 256 / 64, BATCH), 256, 0, stream>>>(
      Ob, NB, LTOT, WeoT, beo, edgeOut, 1, 0, EB, INNER, 256);
}
